// CrossGCNPredictor_28613072126234
// MI455X (gfx1250) — matrix operand order repaired and checked by reading, not compiled, not run on hardware
//
#include <hip/hip_runtime.h>
#include <hip/hip_bf16.h>

// ---------------------------------------------------------------------------
// CrossGCNPredictor for MI455X (gfx1250, wave32, WMMA).
// All activations live in bf16 with K-padded rows; weights pre-packed to
// bf16 [Np][Kp] transposed. GEMMs: v_wmma_f32_16x16x32_bf16, f32 accum,
// unguarded b128 global->LDS staging, register prefetch of next K tile.
// ---------------------------------------------------------------------------

typedef __attribute__((ext_vector_type(16))) __bf16 v16bf;
typedef __attribute__((ext_vector_type(8)))  float  v8f;

union FragAB {
    v16bf v;
    uint4 u[2];
};

#define BM 128
#define BN 64
#define TK 32
#define LDSW 40   // 32 + 8 pad (ushort units); row = 80B, keeps b128 alignment

__device__ __forceinline__ unsigned short f2bf(float f) {
    unsigned u = __float_as_uint(f);
    unsigned r = u + 0x7FFFu + ((u >> 16) & 1u);   // round-to-nearest-even
    return (unsigned short)(r >> 16);
}
__device__ __forceinline__ float bf2f(unsigned short h) {
    return __uint_as_float(((unsigned)h) << 16);
}

// Segment table (compile-time constants of the model)
__constant__ int   c_starts[8] = {0, 41, 70, 91, 110, 130, 137, 158};
__constant__ int   c_ends[8]   = {41, 70, 91, 110, 130, 137, 158, 200};
__constant__ float c_invcnt[8] = {1.f/41.f, 1.f/29.f, 1.f/21.f, 1.f/19.f,
                                  1.f/20.f, 1.f/7.f,  1.f/21.f, 1.f/42.f};

// ---------------------------------------------------------------------------
// GEMM: Y[M,ldy](bf16) = act(X[M,lda](bf16) @ Wp + bias), f32 accumulate.
// X rows padded to lda (mult of 32, zero pad). Wp packed [gridy*64][lda] bf16,
// zero padded. M must be a multiple of 128. Block = 8 waves; block tile
// 128x64; wave tile 32x32 (4 WMMA per K step).
// ---------------------------------------------------------------------------
__global__ __launch_bounds__(256) void gemm_wmma_bf16(
    const unsigned short* __restrict__ X, int lda,
    const unsigned short* __restrict__ Wp,
    const float* __restrict__ bias,
    unsigned short* __restrict__ Y, int ldy,
    int N, int act)
{
    __shared__ alignas(16) unsigned short As[BM * LDSW]; // 10240 B
    __shared__ alignas(16) unsigned short Bs[BN * LDSW]; //  5120 B

    const int tid  = threadIdx.x;
    const int wave = tid >> 5;
    const int lane = tid & 31;
    const int hlf  = lane >> 4;
    const int l15  = lane & 15;

    const int m0 = blockIdx.x * BM;
    const int n0 = blockIdx.y * BN;
    const int wm = wave & 3;      // 32-row band
    const int wn = wave >> 2;     // 32-col band

    // staging map: A = 128x32 (16 bf16/thread), B = 64x32 (8 bf16/thread)
    const int ar = tid >> 1, ak = (tid & 1) * 16;
    const int br = tid >> 2, bk = (tid & 3) * 8;

    const unsigned short* ax = X  + (long)(m0 + ar) * lda + ak;
    const unsigned short* bx = Wp + (long)(n0 + br) * lda + bk;

    uint4 ra0 = *(const uint4*)ax;
    uint4 ra1 = *(const uint4*)(ax + 8);
    uint4 rb0 = *(const uint4*)bx;

    v8f c00 = {}, c01 = {}, c10 = {}, c11 = {};

    for (int k0 = 0; k0 < lda; k0 += TK) {
        __syncthreads();
        *(uint4*)&As[ar * LDSW + ak]     = ra0;
        *(uint4*)&As[ar * LDSW + ak + 8] = ra1;
        *(uint4*)&Bs[br * LDSW + bk]     = rb0;
        if (k0 + TK < lda) {              // prefetch next tile into registers
            ax += TK; bx += TK;
            ra0 = *(const uint4*)ax;
            ra1 = *(const uint4*)(ax + 8);
            rb0 = *(const uint4*)bx;
        }
        __syncthreads();

        // A frag (16x32): lane=row, lane-half selects +8 in each 16-K block.
        FragAB a0, a1, b0, b1;
        const unsigned short* pa0 = &As[(wm * 32 + l15) * LDSW + hlf * 8];
        a0.u[0] = *(const uint4*)pa0;
        a0.u[1] = *(const uint4*)(pa0 + 16);
        const unsigned short* pa1 = pa0 + 16 * LDSW;
        a1.u[0] = *(const uint4*)pa1;
        a1.u[1] = *(const uint4*)(pa1 + 16);
        // B frag (32x16): lane=col, lane-half selects +8 in each 16-K block.
        const unsigned short* pb0 = &Bs[(wn * 32 + l15) * LDSW + hlf * 8];
        b0.u[0] = *(const uint4*)pb0;
        b0.u[1] = *(const uint4*)(pb0 + 16);
        const unsigned short* pb1 = pb0 + 16 * LDSW;
        b1.u[0] = *(const uint4*)pb1;
        b1.u[1] = *(const uint4*)(pb1 + 16);

        c00 = __builtin_amdgcn_wmma_f32_16x16x32_bf16(false, a0.v, false, b0.v, (short)0, c00, false, false);
        c01 = __builtin_amdgcn_wmma_f32_16x16x32_bf16(false, a0.v, false, b1.v, (short)0, c01, false, false);
        c10 = __builtin_amdgcn_wmma_f32_16x16x32_bf16(false, a1.v, false, b0.v, (short)0, c10, false, false);
        c11 = __builtin_amdgcn_wmma_f32_16x16x32_bf16(false, a1.v, false, b1.v, (short)0, c11, false, false);
    }

    // Epilogue: C layout -> lane l: n=l&15, VGPR r holds m = r + 8*(l>>4).
#pragma unroll
    for (int s = 0; s < 2; ++s) {
#pragma unroll
        for (int t = 0; t < 2; ++t) {
            v8f c = (s == 0) ? (t == 0 ? c00 : c01) : (t == 0 ? c10 : c11);
            int nn = n0 + wn * 32 + t * 16 + l15;
            if (nn < ldy) {
                float bv = (nn < N) ? bias[nn] : 0.f;
                int mb = m0 + wm * 32 + s * 16 + 8 * hlf;
#pragma unroll
                for (int r = 0; r < 8; ++r) {
                    float v = c[r] + bv;
                    if (act) v = v > 0.f ? v : 0.2f * v;
                    if (nn >= N) v = 0.f;          // zero-fill K padding
                    Y[(long)(mb + r) * ldy + nn] = f2bf(v);
                }
            }
        }
    }
}

// ---------------------------------------------------------------------------
// Weight pack: Wp[n*Kp + k] = bf16(W[k*N + n]), zero padded to [Np][Kp].
// ---------------------------------------------------------------------------
__global__ void pack_w_kernel(const float* __restrict__ W,
                              unsigned short* __restrict__ Wp,
                              int K, int N, int Kp, int Np)
{
    int i = blockIdx.x * 256 + threadIdx.x;
    if (i >= Np * Kp) return;
    int n = i / Kp, k = i - n * Kp;
    float v = (n < N && k < K) ? W[(long)k * N + n] : 0.f;
    Wp[i] = f2bf(v);
}

// ---------------------------------------------------------------------------
// Row-sum of f32 matrix: one wave32 per row, shuffle reduction.
// ---------------------------------------------------------------------------
__global__ __launch_bounds__(256) void rowsum_kernel(
    const float* __restrict__ A, float* __restrict__ out, int rows, int cols)
{
    int wave = threadIdx.x >> 5;
    int lane = threadIdx.x & 31;
    int row  = blockIdx.x * 8 + wave;
    if (row >= rows) return;
    const float* p = A + (long)row * cols;
    float s = 0.f;
    for (int c = lane; c < cols; c += 32) s += p[c];
#pragma unroll
    for (int off = 16; off; off >>= 1) s += __shfl_xor(s, off, 32);
    if (lane == 0) out[row] = s;
}

// ---------------------------------------------------------------------------
// _block: y = 0.5*(rs_intra*x + rs_inter[seg]*segmean(x)[seg]).
// IN_F32: X is f32 stride 200 (node_features); else bf16 stride 224.
// Output bf16 stride 224 with zeroed pad columns.
// ---------------------------------------------------------------------------
template <bool IN_F32>
__global__ __launch_bounds__(256) void block_kernel(
    const void* __restrict__ Xv, const float* __restrict__ rs_intra,
    const float* __restrict__ rs_inter, unsigned short* __restrict__ Y)
{
    const int b = blockIdx.x;
    const int f = threadIdx.x;
    if (f >= 224) return;
    unsigned short* yb = Y + (long)b * 200 * 224 + f;
    if (f >= 200) {                 // pad columns
        for (int r = 0; r < 200; ++r) yb[r * 224] = 0;
        return;
    }
    const float*          xf = IN_F32 ? (const float*)Xv + (long)b * 200 * 200 + f : nullptr;
    const unsigned short* xh = IN_F32 ? nullptr : (const unsigned short*)Xv + (long)b * 200 * 224 + f;

    float seg[8];
#pragma unroll
    for (int s = 0; s < 8; ++s) {
        float acc = 0.f;
        for (int r = c_starts[s]; r < c_ends[s]; ++r)
            acc += IN_F32 ? xf[r * 200] : bf2f(xh[r * 224]);
        seg[s] = acc * c_invcnt[s] * rs_inter[b * 8 + s];
    }
    const float* ri = rs_intra + b * 200;
#pragma unroll
    for (int s = 0; s < 8; ++s) {
        float p = seg[s];
        for (int r = c_starts[s]; r < c_ends[s]; ++r) {
            float x = IN_F32 ? xf[r * 200] : bf2f(xh[r * 224]);
            yb[r * 224] = f2bf(0.5f * (ri[r] * x + p));
        }
    }
}

// ---------------------------------------------------------------------------
// bn2d over rows: atomic two-pass stats, then apply (bf16 in/out, stride 224)
// ---------------------------------------------------------------------------
__global__ void zero_kernel(float* p, int n)
{
    int i = blockIdx.x * blockDim.x + threadIdx.x;
    if (i < n) p[i] = 0.f;
}

__global__ __launch_bounds__(256) void bnstats_kernel(
    const unsigned short* __restrict__ X, float* __restrict__ stats,
    long M, int rowsPerBlock)
{
    int c = threadIdx.x;
    if (c >= 200) return;
    long r0 = (long)blockIdx.x * rowsPerBlock;
    long r1 = r0 + rowsPerBlock;
    if (r1 > M) r1 = M;
    float s = 0.f, ss = 0.f;
    for (long r = r0; r < r1; ++r) {
        float v = bf2f(X[r * 224 + c]);
        s += v;
        ss += v * v;
    }
    atomicAdd(&stats[c], s);
    atomicAdd(&stats[200 + c], ss);
}

__global__ __launch_bounds__(256) void bnapply_kernel(
    const unsigned short* __restrict__ X, const float* __restrict__ stats,
    const float* __restrict__ g, const float* __restrict__ be,
    unsigned short* __restrict__ Y, long M)
{
    long r = blockIdx.x;
    int c = threadIdx.x;
    if (c >= 224) return;
    unsigned short o = 0;
    if (c < 200) {
        float mean = stats[c] / (float)M;
        float var  = stats[200 + c] / (float)M - mean * mean;
        float v = (bf2f(X[r * 224 + c]) - mean) * rsqrtf(var + 1e-5f) * g[c] + be[c];
        o = f2bf(v);
    }
    Y[r * 224 + c] = o;
}

// ---------------------------------------------------------------------------
// bn3d: per-ROI channel stats over (B,8)=8192 values, fused apply.
// X: (B,200,8) bf16 contiguous -> Y same (viewed as (1024,1600) for Wc1).
// ---------------------------------------------------------------------------
__global__ __launch_bounds__(256) void bn3_kernel(
    const unsigned short* __restrict__ X, const float* __restrict__ g,
    const float* __restrict__ be, unsigned short* __restrict__ Y)
{
    int r = blockIdx.x;
    __shared__ float red0[256];
    __shared__ float red1[256];
    float s = 0.f, ss = 0.f;
    for (int i = threadIdx.x; i < 1024 * 8; i += 256) {
        int b = i >> 3, j = i & 7;
        float v = bf2f(X[(long)b * 1600 + r * 8 + j]);
        s += v;
        ss += v * v;
    }
    red0[threadIdx.x] = s;
    red1[threadIdx.x] = ss;
    __syncthreads();
    for (int off = 128; off; off >>= 1) {
        if (threadIdx.x < off) {
            red0[threadIdx.x] += red0[threadIdx.x + off];
            red1[threadIdx.x] += red1[threadIdx.x + off];
        }
        __syncthreads();
    }
    float mean = red0[0] / 8192.f;
    float var  = red1[0] / 8192.f - mean * mean;
    float sc = rsqrtf(var + 1e-5f) * g[r];
    float bb = be[r];
    for (int i = threadIdx.x; i < 1024 * 8; i += 256) {
        int b = i >> 3, j = i & 7;
        long idx = (long)b * 1600 + r * 8 + j;
        Y[idx] = f2bf((bf2f(X[idx]) - mean) * sc + bb);
    }
}

// Final tiny GEMM: (1024,32)bf16 @ (32,2)f32 + bc3 -> d_out (1024,2) f32
__global__ void final_kernel(const unsigned short* __restrict__ X,
                             const float* __restrict__ W,
                             const float* __restrict__ b,
                             float* __restrict__ Y)
{
    int i = blockIdx.x * blockDim.x + threadIdx.x;
    if (i >= 2048) return;
    int row = i >> 1, col = i & 1;
    float acc = b[col];
#pragma unroll
    for (int k = 0; k < 32; ++k) acc += bf2f(X[row * 32 + k]) * W[k * 2 + col];
    Y[i] = acc;
}

// ---------------------------------------------------------------------------
// Orchestration
// ---------------------------------------------------------------------------
extern "C" void kernel_launch(void* const* d_in, const int* in_sizes, int n_in,
                              void* d_out, int out_size, void* d_ws, size_t ws_size,
                              hipStream_t stream)
{
    (void)in_sizes; (void)n_in; (void)out_size; (void)ws_size;
    const int B = 1024, ROI = 200, M = B * ROI; // 204800

    const float* intra = (const float*)d_in[1];
    const float* inter = (const float*)d_in[2];
    const float* nf    = (const float*)d_in[3];
    const float* W0  = (const float*)d_in[4];
    const float* b0  = (const float*)d_in[5];
    const float* W1  = (const float*)d_in[6];
    const float* b1  = (const float*)d_in[7];
    const float* g1  = (const float*)d_in[8];
    const float* be1 = (const float*)d_in[9];
    const float* W2  = (const float*)d_in[10];
    const float* b2  = (const float*)d_in[11];
    const float* g2  = (const float*)d_in[12];
    const float* be2 = (const float*)d_in[13];
    const float* W3  = (const float*)d_in[14];
    const float* b3  = (const float*)d_in[15];
    const float* W4  = (const float*)d_in[16];
    const float* b4  = (const float*)d_in[17];
    const float* g3  = (const float*)d_in[18];
    const float* be3 = (const float*)d_in[19];
    const float* Wc1 = (const float*)d_in[20];
    const float* bc1 = (const float*)d_in[21];
    const float* Wc2 = (const float*)d_in[22];
    const float* bc2 = (const float*)d_in[23];
    const float* Wc3 = (const float*)d_in[24];
    const float* bc3 = (const float*)d_in[25];

    // --- workspace carve ---
    char* p = (char*)d_ws;
    float* rs_intra = (float*)p; p += 204800L * 4;
    float* rs_inter = (float*)p; p += 8192L * 4;
    float* stats    = (float*)p; p += 512L * 4;
    unsigned short* actA = (unsigned short*)p; p += 45875200L * 2; // 204800 x 224
    unsigned short* actB = (unsigned short*)p; p += 45875200L * 2;
    unsigned short* W0p  = (unsigned short*)p; p += 256L * 224 * 2;
    unsigned short* W1p  = (unsigned short*)p; p += 256L * 224 * 2;
    unsigned short* W2p  = (unsigned short*)p; p += 256L * 224 * 2;
    unsigned short* W3p  = (unsigned short*)p; p += 64L * 224 * 2;
    unsigned short* W4p  = (unsigned short*)p; p += 64L * 64 * 2;
    unsigned short* Wc1p = (unsigned short*)p; p += 256L * 1600 * 2;
    unsigned short* Wc2p = (unsigned short*)p; p += 64L * 256 * 2;

    dim3 blk(256);
    auto pack = [&](const float* W, unsigned short* Wp, int K, int N, int Kp, int Np) {
        int n = Np * Kp;
        pack_w_kernel<<<(n + 255) / 256, blk, 0, stream>>>(W, Wp, K, N, Kp, Np);
    };
    auto gemm = [&](const unsigned short* X, int lda, const unsigned short* Wp,
                    const float* bias, unsigned short* Y, int ldy,
                    int m, int n, int act) {
        dim3 grid(m / BM, ((n > ldy ? n : ldy) + BN - 1) / BN);
        gemm_wmma_bf16<<<grid, blk, 0, stream>>>(X, lda, Wp, bias, Y, ldy, n, act);
    };
    auto bn2d = [&](const unsigned short* X, const float* g, const float* be,
                    unsigned short* Y) {
        zero_kernel<<<2, blk, 0, stream>>>(stats, 512);
        bnstats_kernel<<<(M + 511) / 512, blk, 0, stream>>>(X, stats, M, 512);
        bnapply_kernel<<<M, blk, 0, stream>>>(X, stats, g, be, Y, M);
    };

    // weight packs (Np = grid.y*64 of their consumer GEMM)
    pack(W0,  W0p,  200, 200, 224, 256);
    pack(W1,  W1p,  200, 200, 224, 256);
    pack(W2,  W2p,  200, 200, 224, 256);
    pack(W3,  W3p,  200, 64,  224, 64);
    pack(W4,  W4p,  64,  8,   64,  64);
    pack(Wc1, Wc1p, 1600, 256, 1600, 256);
    pack(Wc2, Wc2p, 256, 32,  256, 64);

    // adjacency row sums
    rowsum_kernel<<<M / 8, blk, 0, stream>>>(intra, rs_intra, M, ROI);
    rowsum_kernel<<<(B * 8 + 7) / 8, blk, 0, stream>>>(inter, rs_inter, B * 8, 8);

    // block -> GEMM(W0, leaky) -> GEMM(W1) -> bn2d
    block_kernel<true><<<B, blk, 0, stream>>>(nf, rs_intra, rs_inter, actA);
    gemm(actA, 224, W0p, b0, actB, 224, M, 200, 1);
    gemm(actB, 224, W1p, b1, actA, 224, M, 200, 0);
    bn2d(actA, g1, be1, actB);

    // block -> GEMM(W2, leaky) -> bn2d
    block_kernel<false><<<B, blk, 0, stream>>>(actB, rs_intra, rs_inter, actA);
    gemm(actA, 224, W2p, b2, actB, 224, M, 200, 1);
    bn2d(actB, g2, be2, actA);

    // block -> GEMM(W3, leaky) -> GEMM(W4, leaky) -> bn3d
    block_kernel<false><<<B, blk, 0, stream>>>(actA, rs_intra, rs_inter, actB);
    gemm(actB, 224, W3p, b3, actA, 64, M, 64, 1);   // (204800, 64)
    gemm(actA, 64,  W4p, b4, actB, 8,  M, 8,  1);   // (204800, 8) == (1024, 1600)
    bn3_kernel<<<ROI, blk, 0, stream>>>(actB, g3, be3, actA);

    // classifier head
    gemm(actA, 1600, Wc1p, bc1, actB, 256, B, 256, 1);
    gemm(actB, 256,  Wc2p, bc2, actA, 32,  B, 32,  1);
    final_kernel<<<8, blk, 0, stream>>>(actA, Wc3, bc3, (float*)d_out);
}
